// DAEGC_58050777973194
// MI455X (gfx1250) — compile-verified, matrix operand order checked by reading, not run on hardware
//
#include <hip/hip_runtime.h>
#include <hip/hip_bf16.h>
#include <math.h>

// ---------------------------------------------------------------------------
// Problem constants (match reference)
// ---------------------------------------------------------------------------
constexpr int   Nn    = 8192;
constexpr int   FIN   = 512;
constexpr int   HIDC  = 256;
constexpr int   EMBC  = 64;
constexpr float ALPHA = 0.2f;
constexpr float NEGINF = -9e15f;

typedef __attribute__((ext_vector_type(16))) __bf16       bf16x16;
typedef __attribute__((ext_vector_type(8)))  float        f32x8;
typedef __attribute__((ext_vector_type(4)))  unsigned int u32x4;

// ---------------------------------------------------------------------------
// Wave helpers (wave32 on gfx1250)
// ---------------------------------------------------------------------------
__device__ inline float wred_sum(float v) {
#pragma unroll
    for (int o = 16; o > 0; o >>= 1) v += __shfl_xor(v, o, 32);
    return v;
}
__device__ inline float wred_max(float v) {
#pragma unroll
    for (int o = 16; o > 0; o >>= 1) v = fmaxf(v, __shfl_xor(v, o, 32));
    return v;
}

// ---------------------------------------------------------------------------
// WMMA fragment load: 16x32 bf16 tile, row-major with leading dim ld.
// ISA layout (16-bit A/B operand): lanes 0-15 hold rows 0-15 with K=kb..kb+7
// in VGPR0-3 and K=kb+16..kb+23 in VGPR4-7, where kb = (lane>>4)*8.
// Both halves are 16B-contiguous -> two global_load_b128 per lane.
// ---------------------------------------------------------------------------
union FragU { bf16x16 f; u32x4 q[2]; };

__device__ inline bf16x16 load_frag(const __bf16* __restrict__ p, int ld) {
    const int lane = threadIdx.x & 31;
    const int r    = lane & 15;
    const int kb   = (lane >> 4) * 8;
    const __bf16* row = p + (size_t)r * ld + kb;
    FragU u;
    u.q[0] = *(const u32x4*)(row);        // K = kb+0 .. kb+7
    u.q[1] = *(const u32x4*)(row + 16);   // K = kb+16 .. kb+23
    return u.f;
}

__device__ inline f32x8 wmma_bf16(bf16x16 a, bf16x16 b, f32x8 c) {
    // v_wmma_f32_16x16x32_bf16  (codegen-confirmed builtin, 8 args)
    return __builtin_amdgcn_wmma_f32_16x16x32_bf16(
        /*neg_a=*/false, a, /*neg_b=*/false, b,
        /*c_mod=*/(short)0, c, /*reuse_a=*/false, /*reuse_b=*/false);
}

// C/D layout: VGPR v -> row (lane>>4)*8 + v, col = lane&15.
// EPI: 0 = identity, 1 = ELU, 2 = sigmoid
template <int EPI>
__device__ inline void store_tile(float* __restrict__ C, int ldc, f32x8 acc) {
    const int lane = threadIdx.x & 31;
    const int n    = lane & 15;
    const int mb   = (lane >> 4) * 8;
#pragma unroll
    for (int v = 0; v < 8; ++v) {
        float x = acc[v];
        if (EPI == 1) x = (x > 0.f) ? x : (__expf(x) - 1.f);
        if (EPI == 2) x = 1.f / (1.f + __expf(-x));
        C[(size_t)(mb + v) * ldc + n] = x;
    }
}

// ---------------------------------------------------------------------------
// NT GEMM: C[M x Nc] = A[M x K] * BT[Nc x K]^T   (bf16 in, f32 accumulate)
// grid = (M/128, Nc/64), block = 256 threads (8 waves).
// Wave w owns a 32x32 output tile: rows blockIdx.x*128 + (w&3)*32 (+0,+16),
// cols blockIdx.y*64 + (w>>2)*32 (+0,+16).
// Per k-step: 4 fragment loads -> 4 WMMAs (1.0 loads/WMMA, 2x register reuse).
// ---------------------------------------------------------------------------
template <int EPI>
__global__ __launch_bounds__(256)
void gemm_bf16_nt(const __bf16* __restrict__ A, const __bf16* __restrict__ BT,
                  float* __restrict__ C, int M, int Nc, int K) {
    const int w   = threadIdx.x >> 5;
    const int row = blockIdx.x * 128 + (w & 3) * 32;
    const int col = blockIdx.y * 64 + (w >> 2) * 32;

    const __bf16* A0 = A  + (size_t)row * K;
    const __bf16* A1 = A  + (size_t)(row + 16) * K;
    const __bf16* B0 = BT + (size_t)col * K;
    const __bf16* B1 = BT + (size_t)(col + 16) * K;

    f32x8 acc00 = {}, acc01 = {}, acc10 = {}, acc11 = {};
    for (int k = 0; k < K; k += 32) {
        // Speculative streaming prefetch a few steps ahead (global_prefetch_b8);
        // OOB translation failures are silently dropped per ISA.
        __builtin_prefetch(A0 + k + 512, 0, 1);
        __builtin_prefetch(A1 + k + 512, 0, 1);
        __builtin_prefetch(B0 + k + 512, 0, 1);
        __builtin_prefetch(B1 + k + 512, 0, 1);

        bf16x16 a0 = load_frag(A0 + k, K);
        bf16x16 a1 = load_frag(A1 + k, K);
        bf16x16 b0 = load_frag(B0 + k, K);
        bf16x16 b1 = load_frag(B1 + k, K);
        acc00 = wmma_bf16(a0, b0, acc00);
        acc01 = wmma_bf16(a0, b1, acc01);
        acc10 = wmma_bf16(a1, b0, acc10);
        acc11 = wmma_bf16(a1, b1, acc11);
    }
    float* Cp = C + (size_t)row * Nc + col;
    store_tile<EPI>(Cp,                        Nc, acc00);
    store_tile<EPI>(Cp + 16,                   Nc, acc01);
    store_tile<EPI>(Cp + (size_t)16 * Nc,      Nc, acc10);
    store_tile<EPI>(Cp + (size_t)16 * Nc + 16, Nc, acc11);
}

// ---------------------------------------------------------------------------
// Elementwise f32 -> bf16
// ---------------------------------------------------------------------------
__global__ void cvt_bf16(const float* __restrict__ s, __bf16* __restrict__ d,
                         size_t n) {
    size_t i = (size_t)blockIdx.x * blockDim.x + threadIdx.x;
    if (i < n) d[i] = (__bf16)s[i];
}

// Transpose f32 [R x C] -> bf16 [C x R]
__global__ void transpose_bf16(const float* __restrict__ src,
                               __bf16* __restrict__ dst, int R, int C) {
    size_t idx = (size_t)blockIdx.x * blockDim.x + threadIdx.x;
    if (idx >= (size_t)R * C) return;
    int r = (int)(idx / C);
    int c = (int)(idx % C);
    dst[(size_t)c * R + r] = (__bf16)src[idx];
}

// ---------------------------------------------------------------------------
// Per-row dual dot products: s1[i] = h[i].a1, s2[i] = h[i].a2
// 8 waves per block, one wave per row.
// ---------------------------------------------------------------------------
__global__ __launch_bounds__(256)
void rowdot2(const float* __restrict__ H, const float* __restrict__ a1,
             const float* __restrict__ a2, float* __restrict__ s1,
             float* __restrict__ s2, int n, int d) {
    const int wid = threadIdx.x >> 5, lane = threadIdx.x & 31;
    const int i = blockIdx.x * 8 + wid;
    if (i >= n) return;
    const float* hr = H + (size_t)i * d;
    float x1 = 0.f, x2 = 0.f;
    for (int c = lane; c < d; c += 32) {
        float h = hr[c];
        x1 += h * a1[c];
        x2 += h * a2[c];
    }
    x1 = wred_sum(x1);
    x2 = wred_sum(x2);
    if (lane == 0) { s1[i] = x1; s2[i] = x2; }
}

// ---------------------------------------------------------------------------
// Fused attention-row softmax:
//   e = leaky_relu((s_self[i] + s_neigh[j]) * M[i,j]); masked by adj;
//   attn[i,:] = softmax(e)  -> stored bf16 (halves intermediate traffic).
// One block per row; 32 elements per thread held in registers.
// ---------------------------------------------------------------------------
__global__ __launch_bounds__(256)
void attn_softmax(const float* __restrict__ Mm, const float* __restrict__ adj,
                  const float* __restrict__ ssf, const float* __restrict__ sng,
                  __bf16* __restrict__ attn) {
    constexpr int PER = Nn / 256;  // 32
    const int i = blockIdx.x;
    const float* mrow = Mm  + (size_t)i * Nn;
    const float* arow = adj + (size_t)i * Nn;
    __bf16*      orow = attn + (size_t)i * Nn;
    const float si = ssf[i];
    const int lane = threadIdx.x & 31, wid = threadIdx.x >> 5;

    float ev[PER];
    float mx = -3.4e38f;
#pragma unroll
    for (int t = 0; t < PER; ++t) {
        int j = threadIdx.x + t * 256;
        float e = (si + sng[j]) * mrow[j];
        e = (e > 0.f) ? e : ALPHA * e;
        e = (arow[j] > 0.f) ? e : NEGINF;
        ev[t] = e;
        mx = fmaxf(mx, e);
    }

    __shared__ float sm[8];
    __shared__ float bcast;
    mx = wred_max(mx);
    if (lane == 0) sm[wid] = mx;
    __syncthreads();
    if (threadIdx.x == 0) {
        float m = sm[0];
        for (int q = 1; q < 8; ++q) m = fmaxf(m, sm[q]);
        bcast = m;
    }
    __syncthreads();
    mx = bcast;
    __syncthreads();

    float s = 0.f;
#pragma unroll
    for (int t = 0; t < PER; ++t) {
        ev[t] = __expf(ev[t] - mx);
        s += ev[t];
    }
    s = wred_sum(s);
    if (lane == 0) sm[wid] = s;
    __syncthreads();
    if (threadIdx.x == 0) {
        float tt = 0.f;
        for (int q = 0; q < 8; ++q) tt += sm[q];
        bcast = tt;
    }
    __syncthreads();
    const float inv = 1.f / bcast;
#pragma unroll
    for (int t = 0; t < PER; ++t) {
        int j = threadIdx.x + t * 256;
        orow[j] = (__bf16)(ev[t] * inv);
    }
}

// ---------------------------------------------------------------------------
// Row L2-normalize: z (f32, into d_out) and zb (bf16 workspace for z z^T).
// One wave per row (d = 64).
// ---------------------------------------------------------------------------
__global__ __launch_bounds__(256)
void l2norm(const float* __restrict__ H, float* __restrict__ z,
            __bf16* __restrict__ zb, int n, int d) {
    const int wid = threadIdx.x >> 5, lane = threadIdx.x & 31;
    const int i = blockIdx.x * 8 + wid;
    if (i >= n) return;
    const float* hr = H + (size_t)i * d;
    float s = 0.f;
    for (int c = lane; c < d; c += 32) { float v = hr[c]; s += v * v; }
    s = wred_sum(s);
    const float inv = 1.f / fmaxf(sqrtf(s), 1e-12f);
    for (int c = lane; c < d; c += 32) {
        float v = hr[c] * inv;
        z [(size_t)i * d + c] = v;
        zb[(size_t)i * d + c] = (__bf16)v;
    }
}

// ---------------------------------------------------------------------------
// Host-side orchestration
// ---------------------------------------------------------------------------
extern "C" void kernel_launch(void* const* d_in, const int* in_sizes, int n_in,
                              void* d_out, int out_size, void* d_ws, size_t ws_size,
                              hipStream_t stream) {
    (void)in_sizes; (void)n_in; (void)out_size; (void)ws_size;

    const float* x   = (const float*)d_in[0];
    const float* adj = (const float*)d_in[1];
    const float* Mm  = (const float*)d_in[2];
    const float* W1  = (const float*)d_in[3];
    const float* as1 = (const float*)d_in[4];
    const float* an1 = (const float*)d_in[5];
    const float* W2  = (const float*)d_in[6];
    const float* as2 = (const float*)d_in[7];
    const float* an2 = (const float*)d_in[8];

    float* A_pred = (float*)d_out;                       // [N x N]
    float* z      = (float*)d_out + (size_t)Nn * Nn;     // [N x EMB]

    // attn (bf16, 128 MB) lives in the A_pred region: it is dead until the
    // final z z^T GEMM overwrites it.
    __bf16* attnb = (__bf16*)d_out;

    // Bump allocator over workspace (~38 MB total).
    char* wsp = (char*)d_ws;
    auto alloc = [&](size_t bytes) -> void* {
        void* p = wsp;
        wsp += (bytes + 255) & ~(size_t)255;
        return p;
    };
    __bf16* xb    = (__bf16*)alloc((size_t)Nn * FIN * 2);
    __bf16* w1t   = (__bf16*)alloc((size_t)FIN * HIDC * 2);
    float*  h1f   = (float*) alloc((size_t)Nn * HIDC * 4);
    __bf16* h1t   = (__bf16*)alloc((size_t)Nn * HIDC * 2);
    float*  h1p   = (float*) alloc((size_t)Nn * HIDC * 4);
    __bf16* h1pb  = (__bf16*)alloc((size_t)Nn * HIDC * 2);
    __bf16* w2t   = (__bf16*)alloc((size_t)HIDC * EMBC * 2);
    float*  h2f   = (float*) alloc((size_t)Nn * EMBC * 4);
    __bf16* h2t   = (__bf16*)alloc((size_t)Nn * EMBC * 2);
    float*  h2p   = (float*) alloc((size_t)Nn * EMBC * 4);
    __bf16* zb    = (__bf16*)alloc((size_t)Nn * EMBC * 2);
    float*  ssf   = (float*) alloc((size_t)Nn * 4);
    float*  sng   = (float*) alloc((size_t)Nn * 4);

    auto gridCvt = [](size_t n) { return dim3((unsigned)((n + 255) / 256)); };

    // ---- Layer 1 ----------------------------------------------------------
    cvt_bf16<<<gridCvt((size_t)Nn * FIN), 256, 0, stream>>>(x, xb, (size_t)Nn * FIN);
    transpose_bf16<<<gridCvt((size_t)FIN * HIDC), 256, 0, stream>>>(W1, w1t, FIN, HIDC);
    gemm_bf16_nt<0><<<dim3(Nn / 128, HIDC / 64), 256, 0, stream>>>(xb, w1t, h1f, Nn, HIDC, FIN);

    rowdot2<<<Nn / 8, 256, 0, stream>>>(h1f, as1, an1, ssf, sng, Nn, HIDC);
    attn_softmax<<<Nn, 256, 0, stream>>>(Mm, adj, ssf, sng, attnb);

    transpose_bf16<<<gridCvt((size_t)Nn * HIDC), 256, 0, stream>>>(h1f, h1t, Nn, HIDC);
    gemm_bf16_nt<1><<<dim3(Nn / 128, HIDC / 64), 256, 0, stream>>>(attnb, h1t, h1p, Nn, HIDC, Nn); // + ELU

    // ---- Layer 2 ----------------------------------------------------------
    cvt_bf16<<<gridCvt((size_t)Nn * HIDC), 256, 0, stream>>>(h1p, h1pb, (size_t)Nn * HIDC);
    transpose_bf16<<<gridCvt((size_t)HIDC * EMBC), 256, 0, stream>>>(W2, w2t, HIDC, EMBC);
    gemm_bf16_nt<0><<<dim3(Nn / 128, EMBC / 64), 256, 0, stream>>>(h1pb, w2t, h2f, Nn, EMBC, HIDC);

    rowdot2<<<Nn / 8, 256, 0, stream>>>(h2f, as2, an2, ssf, sng, Nn, EMBC);
    attn_softmax<<<Nn, 256, 0, stream>>>(Mm, adj, ssf, sng, attnb);

    transpose_bf16<<<gridCvt((size_t)Nn * EMBC), 256, 0, stream>>>(h2f, h2t, Nn, EMBC);
    gemm_bf16_nt<1><<<dim3(Nn / 128, EMBC / 64), 256, 0, stream>>>(attnb, h2t, h2p, Nn, EMBC, Nn); // + ELU

    // ---- Normalize + A_pred ----------------------------------------------
    l2norm<<<Nn / 8, 256, 0, stream>>>(h2p, z, zb, Nn, EMBC);
    // A_pred = sigmoid(z z^T): NT GEMM with B^T == z itself; overwrites attn region.
    gemm_bf16_nt<2><<<dim3(Nn / 128, Nn / 64), 256, 0, stream>>>(zb, zb, A_pred, Nn, Nn, EMBC);
}
